// Mamba2Mixer_38774964748419
// MI455X (gfx1250) — matrix operand order repaired and checked by reading, not compiled, not run on hardware
//
#include <hip/hip_runtime.h>
#include <hip/hip_bf16.h>

// ---------------- problem constants ----------------
#define BATCH  2
#define SEQ    2048
#define BLR    (BATCH*SEQ)                 // 4096 rows
#define NE     2048                        // N_EMBD
#define DIN    4096                        // D_INNER
#define NH     64                          // heads
#define HD     64                          // head dim
#define DSTATE 128
#define CONVD  (DIN + 2*DSTATE)            // 4352
#define DIP    (2*DIN + 2*DSTATE + NH)     // 8512
#define QC     256                         // chunk
#define NCH    (SEQ/QC)                    // 8
#define RMSEPS 1.1920929e-07f

// LDS row strides (elements) chosen for alignment + bank-conflict-free gathers
#define SB_STRIDE  136                     // 272 B/row: 16B-aligned for async b128
#define SXT_STRIDE 258
#define SCB_STRIDE 34
#define SG_STRIDE  36                      // 72 B/row: 8B-aligned packed stores

// ---------------- WMMA types ----------------
typedef __attribute__((ext_vector_type(16))) __bf16 v16bf;
typedef __attribute__((ext_vector_type(8)))  float  v8f;

union FragAB { v16bf v; unsigned int u[8]; unsigned short s[16]; };
union FragC  { v8f v; float f[8]; };

#define WMMA_BF16(a,b,c) __builtin_amdgcn_wmma_f32_16x16x32_bf16(false,(a),false,(b),(short)0,(c),false,false)

__device__ inline unsigned short f2bf(float x) {
  union { float f; unsigned int u; } c; c.f = x;
  unsigned int r = c.u + 0x7FFFu + ((c.u >> 16) & 1u);   // round-to-nearest-even
  return (unsigned short)(r >> 16);
}
__device__ inline float bf2f(unsigned short b) {
  union { float f; unsigned int u; } c; c.u = ((unsigned int)b) << 16;
  return c.f;
}
__device__ inline float siluf(float x) { return x / (1.f + __expf(-x)); }
__device__ inline float softplusf(float x) { return (x > 20.f) ? x : log1pf(__expf(x)); }

// =====================================================================
// Generic GEMM: C[M,N] = A[M,K] * B[N,K]^T  (fp32 global, bf16 WMMA math)
// 128x128 tile, K-step 32, 8 waves; wave -> 32x64 (2x4 16x16 WMMA tiles)
// float4 staging loads + next-tile global_prefetch_b8
// =====================================================================
__global__ __launch_bounds__(256) void gemm_f32_bf16(
    const float* __restrict__ A, const float* __restrict__ Bm,
    float* __restrict__ C, int M, int N, int K) {
  __shared__ __align__(16) unsigned short sA[128][SG_STRIDE];
  __shared__ __align__(16) unsigned short sB[128][SG_STRIDE];
  const int tid = threadIdx.x;
  const int lane = tid & 31, wv = tid >> 5;
  const int hf = (lane >> 4) & 1, l15 = lane & 15;
  const int m0 = blockIdx.y * 128, n0 = blockIdx.x * 128;
  const int wr = (wv & 3) * 32, wc = (wv >> 2) * 64;

  FragC acc[2][4];
  for (int i = 0; i < 2; i++) for (int j = 0; j < 4; j++)
    for (int r = 0; r < 8; r++) acc[i][j].f[r] = 0.f;

  for (int kt = 0; kt < K; kt += 32) {
    __syncthreads();
    // stage A tile (128x32) : 1024 float4 segments
    #pragma unroll
    for (int j = 0; j < 4; j++) {
      int i = tid + 256 * j;
      int r = i >> 3, s4 = i & 7;
      int gm = m0 + r;
      float4 v = make_float4(0.f, 0.f, 0.f, 0.f);
      if (gm < M) {
        const float* p = A + (size_t)gm * K + kt + s4 * 4;
        v = *(const float4*)p;
        if (kt + 32 < K) __builtin_prefetch(p + 32, 0, 1);   // global_prefetch_b8
      }
      uint2 pk;
      pk.x = (unsigned)f2bf(v.x) | ((unsigned)f2bf(v.y) << 16);
      pk.y = (unsigned)f2bf(v.z) | ((unsigned)f2bf(v.w) << 16);
      *(uint2*)&sA[r][s4 * 4] = pk;
    }
    // stage B tile (128x32)
    #pragma unroll
    for (int j = 0; j < 4; j++) {
      int i = tid + 256 * j;
      int r = i >> 3, s4 = i & 7;
      int gn = n0 + r;
      float4 v = make_float4(0.f, 0.f, 0.f, 0.f);
      if (gn < N) {
        const float* p = Bm + (size_t)gn * K + kt + s4 * 4;
        v = *(const float4*)p;
        if (kt + 32 < K) __builtin_prefetch(p + 32, 0, 1);
      }
      uint2 pk;
      pk.x = (unsigned)f2bf(v.x) | ((unsigned)f2bf(v.y) << 16);
      pk.y = (unsigned)f2bf(v.z) | ((unsigned)f2bf(v.w) << 16);
      *(uint2*)&sB[r][s4 * 4] = pk;
    }
    __syncthreads();

    FragAB fa[2], fb[4];
    #pragma unroll
    for (int ib = 0; ib < 2; ib++) {
      const unsigned short* p = &sA[wr + ib*16 + l15][0];
      #pragma unroll
      for (int wd = 0; wd < 8; wd++) {
        int k = 2*(wd & 3) + 16*(wd >> 2) + 8*hf;        // A-frag layout
        fa[ib].u[wd] = *(const unsigned int*)&p[k];
      }
    }
    #pragma unroll
    for (int jb = 0; jb < 4; jb++) {
      const unsigned short* p = &sB[wc + jb*16 + l15][0];
      #pragma unroll
      for (int wd = 0; wd < 8; wd++) {
        int k = 2*(wd & 3) + 16*(wd >> 2) + 8*hf;        // B-frag layout
        fb[jb].u[wd] = *(const unsigned int*)&p[k];
      }
    }
    #pragma unroll
    for (int ib = 0; ib < 2; ib++)
      #pragma unroll
      for (int jb = 0; jb < 4; jb++)
        acc[ib][jb].v = WMMA_BF16(fa[ib].v, fb[jb].v, acc[ib][jb].v);
  }

  for (int ib = 0; ib < 2; ib++) for (int jb = 0; jb < 4; jb++)
    #pragma unroll
    for (int r = 0; r < 8; r++) {
      int gm = m0 + wr + ib*16 + r + 8*hf;
      int gn = n0 + wc + jb*16 + l15;
      if (gm < M && gn < N) C[(size_t)gm * N + gn] = acc[ib][jb].f[r];
    }
}

// =====================================================================
// Depthwise causal conv(4) + bias + SiLU on xBC slice of zxbcdt
// =====================================================================
__global__ __launch_bounds__(256) void conv_silu_k(
    const float* __restrict__ zx, const float* __restrict__ cw,
    const float* __restrict__ cb, float* __restrict__ xBC) {
  size_t idx = (size_t)blockIdx.x * 256 + threadIdx.x;
  if (idx >= (size_t)BLR * CONVD) return;
  int ch = (int)(idx % CONVD);
  int bl = (int)(idx / CONVD);
  int b = bl / SEQ, l = bl % SEQ;
  float acc = cb[ch];
  #pragma unroll
  for (int j = 0; j < 4; j++) {
    int ls = l - 3 + j;
    if (ls >= 0)
      acc += zx[((size_t)(b * SEQ + ls)) * DIP + DIN + ch] * cw[ch * 4 + j];
  }
  xBC[idx] = siluf(acc);
}

// =====================================================================
// bf16 prep: Xdt = x*softplus(dt_raw+bias), plus B/C bf16 copies
// =====================================================================
__global__ __launch_bounds__(256) void prep_k(
    const float* __restrict__ zx, const float* __restrict__ xBC,
    const float* __restrict__ dt_bias,
    unsigned short* __restrict__ Xdt, unsigned short* __restrict__ Bbf,
    unsigned short* __restrict__ Cbf) {
  size_t idx = (size_t)blockIdx.x * 256 + threadIdx.x;
  if (idx >= (size_t)BLR * CONVD) return;
  int ch = (int)(idx % CONVD);
  int bl = (int)(idx / CONVD);
  float v = xBC[idx];
  if (ch < DIN) {
    int h = ch >> 6;
    float dtr = zx[(size_t)bl * DIP + DIN + CONVD + h] + dt_bias[h];
    Xdt[(size_t)bl * DIN + ch] = f2bf(v * softplusf(dtr));
  } else if (ch < DIN + DSTATE) {
    Bbf[(size_t)bl * DSTATE + (ch - DIN)] = f2bf(v);
  } else {
    Cbf[(size_t)bl * DSTATE + (ch - DIN - DSTATE)] = f2bf(v);
  }
}

// =====================================================================
// Per (b,h,chunk) inclusive cumsum of g = -exp(A_log)*softplus(dt)
// =====================================================================
__global__ void cumsum_k(const float* __restrict__ zx, const float* __restrict__ A_log,
                         const float* __restrict__ dt_bias, float* __restrict__ acum) {
  int idx = blockIdx.x * blockDim.x + threadIdx.x;
  if (idx >= BATCH * NH * NCH) return;
  int c = idx % NCH, h = (idx / NCH) % NH, b = idx / (NCH * NH);
  float Ac = -__expf(A_log[h]);
  float db = dt_bias[h];
  float s = 0.f;
  float* dst = acum + ((size_t)(b * NH + h) * NCH + c) * QC;
  for (int q = 0; q < QC; q++) {
    int l = c * QC + q;
    float x = zx[((size_t)(b * SEQ + l)) * DIP + DIN + CONVD + h] + db;
    s += Ac * softplusf(x);
    dst[q] = s;
  }
}

// =====================================================================
// SSD intra-chunk: Y_diag = (L o C.B^T).X  and  states = B^T.(decay o X)
// one workgroup per (b, chunk, head); dynamic LDS ~142KB
// B tile staged via GLOBAL_LOAD_ASYNC_TO_LDS_B128 (ASYNCcnt path)
// =====================================================================
__global__ __launch_bounds__(256) void ssd_chunk_k(
    const unsigned short* __restrict__ Xdt, const unsigned short* __restrict__ Bbf,
    const unsigned short* __restrict__ Cbf, const float* __restrict__ acum,
    float* __restrict__ Y, float* __restrict__ states) {
  extern __shared__ char smem[];
  unsigned short* sB   = (unsigned short*)smem;            // [256][136]  B (l,n)
  unsigned short* sXT  = sB  + 256 * SB_STRIDE;            // [64][258]   X^T (p,l)
  unsigned short* sXdT = sXT + 64 * SXT_STRIDE;            // [64][258]   (decay o X)^T
  float*          sac  = (float*)(sXdT + 64 * SXT_STRIDE); // [256]
  unsigned short* sCB  = (unsigned short*)(sac + 256);     // [8 waves][16][34]

  const int tid = threadIdx.x;
  const int lane = tid & 31, wv = tid >> 5;
  const int hf = (lane >> 4) & 1, l15 = lane & 15;
  const int gid = blockIdx.x;
  const int h = gid % NH, c = (gid / NH) % NCH, b = gid / (NH * NCH);
  const int blbase = b * SEQ + c * QC;
  const float* ac = acum + ((size_t)(b * NH + h) * NCH + c) * QC;

  // ---- stage B tile (256 x 128 bf16) via async global->LDS, 16 B per lane ----
  {
    const unsigned short* gB = Bbf + (size_t)blbase * DSTATE;
    #pragma unroll
    for (int j = 0; j < 16; j++) {
      int i = tid + 256 * j;                       // 0..4095 segments
      int r = i >> 4, seg = i & 15;
      unsigned long long gaddr =
          (unsigned long long)(size_t)(gB + (size_t)r * DSTATE + seg * 8);
      unsigned lofs = (unsigned)(size_t)&sB[r * SB_STRIDE + seg * 8];
      asm volatile("global_load_async_to_lds_b128 %0, %1, off"
                   :: "v"(lofs), "v"(gaddr)
                   : "memory");
    }
  }

  sac[tid] = ac[tid];
  // stage X^T and decayed X^T (needs f32 math, stays on VALU path)
  float aclast = ac[QC - 1];
  for (int i = tid; i < 256 * 64; i += 256) {
    int l = i >> 6, p = i & 63;
    unsigned short x = Xdt[((size_t)(blbase + l)) * DIN + h * HD + p];
    sXT[p * SXT_STRIDE + l] = x;
    sXdT[p * SXT_STRIDE + l] = f2bf(bf2f(x) * __expf(aclast - ac[l]));
  }
  asm volatile("s_wait_asynccnt 0x0" ::: "memory");   // async B-tile copies done
  __syncthreads();

  // A-fragments for C (gathered once from global): 2 l-blocks x 4 k-steps
  FragAB fc[2][4];
  #pragma unroll
  for (int lb = 0; lb < 2; lb++) {
    const unsigned int* pC =
        (const unsigned int*)(Cbf + ((size_t)(blbase + wv * 32 + lb * 16 + l15)) * DSTATE);
    #pragma unroll
    for (int kk = 0; kk < 4; kk++)
      #pragma unroll
      for (int wd = 0; wd < 8; wd++) {
        int k = kk * 32 + 2 * (wd & 3) + 16 * (wd >> 2) + 8 * hf;
        fc[lb][kk].u[wd] = pC[k >> 1];
      }
  }

  FragC accY[2][4], accS[4];
  for (int i = 0; i < 2; i++) for (int j = 0; j < 4; j++)
    for (int r = 0; r < 8; r++) accY[i][j].f[r] = 0.f;
  for (int j = 0; j < 4; j++) for (int r = 0; r < 8; r++) accS[j].f[r] = 0.f;

  unsigned short* sCBw = sCB + wv * 16 * SCB_STRIDE;

  for (int sb2 = 0; sb2 < 8; sb2++) {                    // 32-wide s/l blocks
    // B-matrix fragments of X for this k-block (k = l contiguous in sXT rows)
    FragAB fx[4];
    #pragma unroll
    for (int pt = 0; pt < 4; pt++) {
      const unsigned short* px = &sXT[(pt * 16 + l15) * SXT_STRIDE];
      #pragma unroll
      for (int wd = 0; wd < 8; wd++) {
        int l = sb2 * 32 + 2 * (wd & 3) + 16 * (wd >> 2) + 8 * hf;
        fx[pt].u[wd] = *(const unsigned int*)&px[l];
      }
    }
    // ---- states: accS += B^T (A) x (decay o X) (B) ----
    FragAB fbt;
    #pragma unroll
    for (int e = 0; e < 16; e++) {
      int l = sb2 * 32 + (e & 7) + ((e >> 3) << 4) + 8 * hf;
      fbt.s[e] = sB[l * SB_STRIDE + wv * 16 + l15];
    }
    #pragma unroll
    for (int pt = 0; pt < 4; pt++) {
      FragAB fxd;
      const unsigned short* pxd = &sXdT[(pt * 16 + l15) * SXT_STRIDE];
      #pragma unroll
      for (int wd = 0; wd < 8; wd++) {
        int l = sb2 * 32 + 2 * (wd & 3) + 16 * (wd >> 2) + 8 * hf;
        fxd.u[wd] = *(const unsigned int*)&pxd[l];
      }
      accS[pt].v = WMMA_BF16(fbt.v, fxd.v, accS[pt].v);
    }

    // ---- diag: CB tile -> mask/scale -> LDS -> A-frag -> Y += CB.X ----
    for (int lb = 0; lb < 2; lb++) {
      #pragma unroll
      for (int ss = 0; ss < 2; ss++) {
        int sblk = sb2 * 2 + ss;
        FragC cb;
        for (int r = 0; r < 8; r++) cb.f[r] = 0.f;
        #pragma unroll
        for (int kk = 0; kk < 4; kk++) {
          FragAB fbb;
          const unsigned short* pb = &sB[(sblk * 16 + l15) * SB_STRIDE];
          #pragma unroll
          for (int wd = 0; wd < 8; wd++) {
            int n = kk * 32 + 2 * (wd & 3) + 16 * (wd >> 2) + 8 * hf;
            fbb.u[wd] = *(const unsigned int*)&pb[n];
          }
          cb.v = WMMA_BF16(fc[lb][kk].v, fbb.v, cb.v);
        }
        int si = sblk * 16 + l15;
        #pragma unroll
        for (int r = 0; r < 8; r++) {
          int li = wv * 32 + lb * 16 + r + 8 * hf;
          float v = (si <= li) ? cb.f[r] * __expf(sac[li] - sac[si]) : 0.f;
          sCBw[(r + 8 * hf) * SCB_STRIDE + ss * 16 + l15] = f2bf(v);
        }
      }
      __builtin_amdgcn_wave_barrier();
      FragAB fcb;
      {
        const unsigned short* pm = &sCBw[l15 * SCB_STRIDE];
        #pragma unroll
        for (int wd = 0; wd < 8; wd++) {
          int k = 2 * (wd & 3) + 16 * (wd >> 2) + 8 * hf;
          fcb.u[wd] = *(const unsigned int*)&pm[k];
        }
      }
      #pragma unroll
      for (int pt = 0; pt < 4; pt++)
        accY[lb][pt].v = WMMA_BF16(fcb.v, fx[pt].v, accY[lb][pt].v);
      __builtin_amdgcn_wave_barrier();
    }
  }

  // store Y_diag
  for (int lb = 0; lb < 2; lb++) for (int pt = 0; pt < 4; pt++)
    #pragma unroll
    for (int r = 0; r < 8; r++) {
      int li = wv * 32 + lb * 16 + r + 8 * hf;
      int p  = pt * 16 + l15;
      Y[((size_t)(blbase + li)) * DIN + h * HD + p] = accY[lb][pt].f[r];
    }
  // store states: layout [b][c][h][p][n]
  size_t sb0 = ((size_t)b * NCH + c) * NH + h;
  for (int pt = 0; pt < 4; pt++)
    #pragma unroll
    for (int r = 0; r < 8; r++) {
      int n = wv * 16 + r + 8 * hf;
      int p = pt * 16 + l15;
      states[(sb0 * HD + p) * DSTATE + n] = accS[pt].f[r];
    }
}

// =====================================================================
// Inter-chunk recurrence: sprev[c] = state entering chunk c
// =====================================================================
__global__ __launch_bounds__(256) void chain_k(
    const float* __restrict__ acum, const float* __restrict__ states,
    float* __restrict__ sprev) {
  int idx = blockIdx.x * 256 + threadIdx.x;
  int n = idx & 127;
  int p = (idx >> 7) & 63;
  int h = (idx >> 13) & 63;
  int b = idx >> 19;
  float run = 0.f;
  for (int c = 0; c < NCH; c++) {
    size_t off = ((((size_t)b * NCH + c) * NH + h) * HD + p) * DSTATE + n;
    sprev[off] = run;
    float cd = __expf(acum[((size_t)(b * NH + h) * NCH + c) * QC + (QC - 1)]);
    run = run * cd + states[off];
  }
}

// =====================================================================
// Y_off = exp(acum[l]) * (C . sprev)  +  skip x*D; accumulate into Y
// =====================================================================
__global__ __launch_bounds__(256) void yoff_k(
    const unsigned short* __restrict__ Cbf, const float* __restrict__ sprev,
    const float* __restrict__ acum, const float* __restrict__ xBC,
    const float* __restrict__ Dv, float* __restrict__ Y) {
  __shared__ __align__(16) unsigned short sST[64 * 130];   // sprev [p][n]
  __shared__ float sac[256];
  const int tid = threadIdx.x;
  const int lane = tid & 31, wv = tid >> 5;
  const int hf = (lane >> 4) & 1, l15 = lane & 15;
  const int gid = blockIdx.x;
  const int h = gid % NH, c = (gid / NH) % NCH, b = gid / (NH * NCH);
  const int blbase = b * SEQ + c * QC;
  const float* ac = acum + ((size_t)(b * NH + h) * NCH + c) * QC;
  sac[tid] = ac[tid];
  size_t spb = (((size_t)b * NCH + c) * NH + h) * HD * DSTATE;
  for (int i = tid; i < 64 * 128; i += 256) {
    int p = i >> 7, n = i & 127;
    sST[p * 130 + n] = f2bf(sprev[spb + (size_t)p * DSTATE + n]);
  }
  __syncthreads();

  FragAB fcg[2][4];
  #pragma unroll
  for (int lb = 0; lb < 2; lb++) {
    const unsigned int* pC =
        (const unsigned int*)(Cbf + ((size_t)(blbase + wv * 32 + lb * 16 + l15)) * DSTATE);
    #pragma unroll
    for (int kk = 0; kk < 4; kk++)
      #pragma unroll
      for (int wd = 0; wd < 8; wd++) {
        int k = kk * 32 + 2 * (wd & 3) + 16 * (wd >> 2) + 8 * hf;
        fcg[lb][kk].u[wd] = pC[k >> 1];
      }
  }

  FragC acc[2][4];
  for (int i = 0; i < 2; i++) for (int j = 0; j < 4; j++)
    for (int r = 0; r < 8; r++) acc[i][j].f[r] = 0.f;

  for (int kk = 0; kk < 4; kk++) {
    FragAB fs[4];
    #pragma unroll
    for (int pt = 0; pt < 4; pt++) {
      const unsigned short* ps = &sST[(pt * 16 + l15) * 130];
      #pragma unroll
      for (int wd = 0; wd < 8; wd++) {
        int n = kk * 32 + 2 * (wd & 3) + 16 * (wd >> 2) + 8 * hf;
        fs[pt].u[wd] = *(const unsigned int*)&ps[n];
      }
    }
    #pragma unroll
    for (int lb = 0; lb < 2; lb++)
      #pragma unroll
      for (int pt = 0; pt < 4; pt++)
        acc[lb][pt].v = WMMA_BF16(fcg[lb][kk].v, fs[pt].v, acc[lb][pt].v);
  }

  float Dh = Dv[h];
  for (int lb = 0; lb < 2; lb++) for (int pt = 0; pt < 4; pt++)
    #pragma unroll
    for (int r = 0; r < 8; r++) {
      int li = wv * 32 + lb * 16 + r + 8 * hf;
      int p  = pt * 16 + l15;
      size_t row = (size_t)(blbase + li);
      float sdo = __expf(sac[li]);
      float x = xBC[row * CONVD + h * HD + p];
      size_t yi = row * DIN + h * HD + p;
      Y[yi] = Y[yi] + sdo * acc[lb][pt].f[r] + x * Dh;
    }
}

// =====================================================================
// RMS-norm over D_INNER + norm_w * silu(z) gate (in place on Y)
// =====================================================================
__global__ __launch_bounds__(256) void rmsgate_k(
    float* __restrict__ Y, const float* __restrict__ zx,
    const float* __restrict__ nw) {
  __shared__ float red[256];
  int row = blockIdx.x, tid = threadIdx.x;
  float ss = 0.f;
  for (int i = tid; i < DIN; i += 256) {
    float v = Y[(size_t)row * DIN + i];
    ss += v * v;
  }
  red[tid] = ss;
  __syncthreads();
  for (int s = 128; s > 0; s >>= 1) {
    if (tid < s) red[tid] += red[tid + s];
    __syncthreads();
  }
  float scale = rsqrtf(red[0] / (float)DIN + RMSEPS);
  for (int i = tid; i < DIN; i += 256) {
    float z = zx[(size_t)row * DIP + i];
    float v = Y[(size_t)row * DIN + i] * scale * nw[i];
    Y[(size_t)row * DIN + i] = v * siluf(z);
  }
}

// =====================================================================
extern "C" void kernel_launch(void* const* d_in, const int* in_sizes, int n_in,
                              void* d_out, int out_size, void* d_ws, size_t ws_size,
                              hipStream_t stream) {
  const float* u       = (const float*)d_in[0];
  const float* W_in    = (const float*)d_in[1];
  const float* conv_w  = (const float*)d_in[2];
  const float* conv_b  = (const float*)d_in[3];
  const float* A_log   = (const float*)d_in[4];
  const float* Dv      = (const float*)d_in[5];
  const float* dt_bias = (const float*)d_in[6];
  const float* norm_w  = (const float*)d_in[7];
  const float* W_out   = (const float*)d_in[8];
  float* out = (float*)d_out;

  char* ws = (char*)d_ws;
  size_t off = 0;
  auto alloc = [&](size_t bytes) -> void* {
    void* p = ws + off;
    off += (bytes + 255) & ~(size_t)255;
    return p;
  };
  float* zx              = (float*)alloc((size_t)BLR * DIP * 4);      // 139.5 MB
  float* xBC             = (float*)alloc((size_t)BLR * CONVD * 4);    //  71.3 MB
  float* acum            = (float*)alloc((size_t)BATCH * NH * NCH * QC * 4);
  unsigned short* Xdt    = (unsigned short*)alloc((size_t)BLR * DIN * 2);
  unsigned short* Bbf    = (unsigned short*)alloc((size_t)BLR * DSTATE * 2);
  unsigned short* Cbf    = (unsigned short*)alloc((size_t)BLR * DSTATE * 2);
  float* Y               = (float*)alloc((size_t)BLR * DIN * 4);      //  67.1 MB
  float* states          = (float*)alloc((size_t)BATCH * NCH * NH * HD * DSTATE * 4);
  float* sprev           = (float*)alloc((size_t)BATCH * NCH * NH * HD * DSTATE * 4);
  (void)ws_size; (void)in_sizes; (void)n_in; (void)out_size;

  // 1) zxbcdt = u @ W_in^T                        (M=4096, N=8512, K=2048)
  gemm_f32_bf16<<<dim3((DIP + 127) / 128, (BLR + 127) / 128), 256, 0, stream>>>(
      u, W_in, zx, BLR, DIP, NE);

  // 2) depthwise conv + bias + silu
  {
    size_t tot = (size_t)BLR * CONVD;
    conv_silu_k<<<(unsigned)((tot + 255) / 256), 256, 0, stream>>>(zx, conv_w, conv_b, xBC);
  }
  // 3) bf16 prep: x*dt, B, C
  {
    size_t tot = (size_t)BLR * CONVD;
    prep_k<<<(unsigned)((tot + 255) / 256), 256, 0, stream>>>(zx, xBC, dt_bias, Xdt, Bbf, Cbf);
  }
  // 4) cumsum of A*dt per (b,h,chunk)
  cumsum_k<<<(BATCH * NH * NCH + 255) / 256, 256, 0, stream>>>(zx, A_log, dt_bias, acum);

  // 5) intra-chunk SSD (Y_diag + per-chunk states)
  {
    size_t smemC = (size_t)(256 * SB_STRIDE + 2 * 64 * SXT_STRIDE + 8 * 16 * SCB_STRIDE) * 2
                 + 256 * 4;   // ~145 KB (within CDNA5's 320 KB WGP LDS)
    ssd_chunk_k<<<BATCH * NCH * NH, 256, smemC, stream>>>(Xdt, Bbf, Cbf, acum, Y, states);
  }
  // 6) inter-chunk recurrence
  chain_k<<<(BATCH * NH * HD * DSTATE) / 256, 256, 0, stream>>>(acum, states, sprev);

  // 7) Y += exp(acum)*(C.sprev) + x*D
  yoff_k<<<BATCH * NCH * NH, 256, 0, stream>>>(Cbf, sprev, acum, xBC, Dv, Y);

  // 8) RMS-norm + gate (in place)
  rmsgate_k<<<BLR, 256, 0, stream>>>(Y, zx, norm_w);

  // 9) out = Y @ W_out^T                          (M=4096, N=2048, K=4096)
  gemm_f32_bf16<<<dim3((NE + 127) / 128, (BLR + 127) / 128), 256, 0, stream>>>(
      Y, W_out, out, BLR, NE, DIN);
}
